// EquivariantPooling_27891517620927
// MI455X (gfx1250) — compile-verified, matrix operand order checked by reading, not run on hardware
//
#include <hip/hip_runtime.h>
#include <hip/hip_bf16.h>
#include <math.h>

typedef __attribute__((ext_vector_type(16))) __bf16 v16bf;
typedef __attribute__((ext_vector_type(8)))  float  v8f;

#define H_DIM 256
#define HH_DIM 128
#define NSEG 1024
#define B_ELEMS (2 * 8 * 8 * 32 * 16)   // 65536 bf16 = 128 KB

static __device__ inline v8f wmma_bf16(v16bf a, v16bf b, v8f c) {
    // D = A(16x32 bf16) x B(32x16 bf16) + C(16x16 f32)
    return __builtin_amdgcn_wmma_f32_16x16x32_bf16(
        /*neg_a=*/false, a, /*neg_b=*/false, b,
        /*c_mod=*/(short)0, c, /*reuse_a=*/false, /*reuse_b=*/false);
}

// ---------------------------------------------------------------------------
// Kernel 0: repack gw1/lw1 (fp32 row-major 256x128) into bf16 WMMA-B fragments.
// Bbuf index = (((set*8 + kt)*8 + c)*32 + lane)*16 + j
//   K  = kt*32 + (lane>=16 ? 8 : 0) + (j&7) + (j>=8 ? 16 : 0)
//   N  = c*16 + (lane&15)
// ---------------------------------------------------------------------------
__global__ void prep_weights(const float* __restrict__ gw1,
                             const float* __restrict__ lw1,
                             __bf16* __restrict__ Bbuf) {
    int idx = blockIdx.x * blockDim.x + threadIdx.x;
    if (idx >= B_ELEMS) return;
    int j    =  idx        & 15;
    int lane = (idx >> 4)  & 31;
    int c    = (idx >> 9)  & 7;
    int kt   = (idx >> 12) & 7;
    int set  = (idx >> 15) & 1;
    int K = kt * 32 + ((lane >> 4) * 8) + (j & 7) + ((j >> 3) * 16);
    int N = c * 16 + (lane & 15);
    const float* w = set ? lw1 : gw1;
    Bbuf[idx] = (__bf16)w[K * HH_DIM + N];
}

// ---------------------------------------------------------------------------
// Kernel 1: segment boundaries via binary search over sorted int64 batch.
// ---------------------------------------------------------------------------
__global__ void seg_bounds(const long long* __restrict__ batch,
                           int* __restrict__ seg_start, int n, int nseg) {
    int b = blockIdx.x * blockDim.x + threadIdx.x;
    if (b > nseg) return;
    int lo = 0, hi = n;
    while (lo < hi) {
        int mid = (lo + hi) >> 1;
        if (batch[mid] < (long long)b) lo = mid + 1; else hi = mid;
    }
    seg_start[b] = lo;
}

// ---------------------------------------------------------------------------
// Kernel 2: per-node gate scores via bf16 WMMA.
// Weights staged ONCE per block in LDS (128 KB of 320 KB), waves grid-stride
// over 16-row tiles so B-fragment reads come from LDS (ds_load_b128), not L2.
// ---------------------------------------------------------------------------
__global__ __launch_bounds__(256) void scores_kernel(
    const float* __restrict__ x, const __bf16* __restrict__ Bbuf,
    const float* __restrict__ gb1, const float* __restrict__ gw2,
    const float* __restrict__ gb2, const float* __restrict__ lb1,
    const float* __restrict__ lw2, const float* __restrict__ lb2,
    float* __restrict__ s_g, float* __restrict__ s_l,
    int n, int ntiles) {
    __shared__ __align__(128) __bf16 Bs[B_ELEMS];   // 128 KB LDS

    // Cooperative stage: 256 threads x 512 B each, b128 moves
    {
        const uint4* src = reinterpret_cast<const uint4*>(Bbuf);
        uint4*       dst = reinterpret_cast<uint4*>(Bs);
        for (int i = threadIdx.x; i < B_ELEMS * 2 / 16; i += 256)
            dst[i] = src[i];
    }
    __syncthreads();

    int lane  = threadIdx.x & 31;
    int row16 = lane & 15;
    int half  = lane >> 4;
    int waveInBlock = threadIdx.x >> 5;
    int totalWaves  = gridDim.x * 8;

    float bg2 = gb2[0], bl2 = lb2[0];

    for (int tile = blockIdx.x * 8 + waveInBlock; tile < ntiles; tile += totalWaves) {
        int row = tile * 16 + row16;
        if (row >= n) row = n - 1;                 // clamp tail loads
        const float* xrow = x + (size_t)row * H_DIM;

        v8f zero = {0.f, 0.f, 0.f, 0.f, 0.f, 0.f, 0.f, 0.f};
        v8f accg[8], accl[8];
#pragma unroll
        for (int c = 0; c < 8; c++) { accg[c] = zero; accl[c] = zero; }

        for (int kt = 0; kt < 8; kt++) {
            int kb = kt * 32 + half * 8;           // ISA A-layout K base
            const float4* p0 = reinterpret_cast<const float4*>(xrow + kb);
            const float4* p1 = reinterpret_cast<const float4*>(xrow + kb + 16);
            float4 f0 = p0[0], f1 = p0[1], f2 = p1[0], f3 = p1[1];
            v16bf a;
            a[0]  = (__bf16)f0.x; a[1]  = (__bf16)f0.y; a[2]  = (__bf16)f0.z; a[3]  = (__bf16)f0.w;
            a[4]  = (__bf16)f1.x; a[5]  = (__bf16)f1.y; a[6]  = (__bf16)f1.z; a[7]  = (__bf16)f1.w;
            a[8]  = (__bf16)f2.x; a[9]  = (__bf16)f2.y; a[10] = (__bf16)f2.z; a[11] = (__bf16)f2.w;
            a[12] = (__bf16)f3.x; a[13] = (__bf16)f3.y; a[14] = (__bf16)f3.z; a[15] = (__bf16)f3.w;

            const __bf16* bgbase = Bs + ((size_t)(kt * 8) * 32 + lane) * 16;
            const __bf16* blbase = Bs + ((size_t)((8 + kt) * 8) * 32 + lane) * 16;
#pragma unroll
            for (int c = 0; c < 8; c++) {
                v16bf bg = *reinterpret_cast<const v16bf*>(bgbase + (size_t)c * 512);
                accg[c] = wmma_bf16(a, bg, accg[c]);
                v16bf bl = *reinterpret_cast<const v16bf*>(blbase + (size_t)c * 512);
                accl[c] = wmma_bf16(a, bl, accl[c]);
            }
        }

        // Epilogue: s[M] = sum_N tanh(acc[M][N] + b1[N]) * w2[N] + b2
        // C/D layout: VGPR v = row M=v (lanes 0-15) / M=v+8 (lanes 16-31),
        // N = c*16 + (lane&15).
        float rsg[8] = {0,0,0,0,0,0,0,0};
        float rsl[8] = {0,0,0,0,0,0,0,0};
#pragma unroll
        for (int c = 0; c < 8; c++) {
            int col = c * 16 + row16;
            float b1g = gb1[col], w2g = gw2[col];
            float b1l = lb1[col], w2l = lw2[col];
#pragma unroll
            for (int v = 0; v < 8; v++) {
                rsg[v] += tanhf(accg[c][v] + b1g) * w2g;
                rsl[v] += tanhf(accl[c][v] + b1l) * w2l;
            }
        }
#pragma unroll
        for (int v = 0; v < 8; v++) {
#pragma unroll
            for (int m = 1; m < 16; m <<= 1) {     // reduce within 16-lane half
                rsg[v] += __shfl_xor(rsg[v], m, 32);
                rsl[v] += __shfl_xor(rsl[v], m, 32);
            }
        }
        if (row16 == 0) {
            int base = tile * 16 + half * 8;       // lanes 0 and 16 write 8 rows
#pragma unroll
            for (int v = 0; v < 8; v++) {
                int r = base + v;
                if (r < n) { s_g[r] = rsg[v] + bg2; s_l[r] = rsl[v] + bl2; }
            }
        }
    }
}

// ---------------------------------------------------------------------------
// Kernel 3: per-segment softmax + weighted pooling + output GEMV.
// One 256-thread block per segment. Pool pass uses b128 loads: thread t owns
// columns 4*(t&63)..+3, row-subgroup r = t>>6 covers 4 nodes in parallel.
// ---------------------------------------------------------------------------
__device__ inline float blk_reduce_max(float v, float* red) {
    int t = threadIdx.x;
    red[t] = v; __syncthreads();
    for (int s = 128; s > 0; s >>= 1) {
        if (t < s) red[t] = fmaxf(red[t], red[t + s]);
        __syncthreads();
    }
    float r = red[0]; __syncthreads();
    return r;
}
__device__ inline float blk_reduce_sum(float v, float* red) {
    int t = threadIdx.x;
    red[t] = v; __syncthreads();
    for (int s = 128; s > 0; s >>= 1) {
        if (t < s) red[t] += red[t + s];
        __syncthreads();
    }
    float r = red[0]; __syncthreads();
    return r;
}

__global__ __launch_bounds__(256) void pool_kernel(
    const float* __restrict__ x, const unsigned char* __restrict__ mask,
    const float* __restrict__ s_g, const float* __restrict__ s_l,
    const int* __restrict__ seg_start, const float* __restrict__ ow,
    const float* __restrict__ ob, float* __restrict__ out) {
    __shared__ float red[256];
    __shared__ float eg[256], el[256];
    __shared__ float part[4][512];
    __shared__ float pooled[512];

    int seg = blockIdx.x;
    int i0 = seg_start[seg], i1 = seg_start[seg + 1];
    int t = threadIdx.x;

    // Pass 1: segment maxima (clamped at 0, as in reference)
    float mg = -INFINITY, ml = -INFINITY;
    for (int i = i0 + t; i < i1; i += 256) {
        mg = fmaxf(mg, s_g[i]);
        ml = fmaxf(ml, mask[i] ? s_l[i] : -INFINITY);
    }
    mg = fmaxf(blk_reduce_max(mg, red), 0.f);
    ml = fmaxf(blk_reduce_max(ml, red), 0.f);

    // Pass 2: partition sums
    float zg = 0.f, zl = 0.f;
    for (int i = i0 + t; i < i1; i += 256) {
        zg += __expf(s_g[i] - mg);
        zl += mask[i] ? __expf(s_l[i] - ml) : 0.f;
    }
    zg = blk_reduce_sum(zg, red);
    zl = blk_reduce_sum(zl, red);
    float wg = 1.f / (zg + 1e-8f);
    float wl = 1.f / (zl + 1e-8f);

    // Pass 3: weighted pooling with float4 (global_load_b128) x-streaming
    int r  = t >> 6;            // 0..3 row subgroup
    int c4 = (t & 63) * 4;      // 4 contiguous columns
    float4 pg4 = {0.f, 0.f, 0.f, 0.f};
    float4 pl4 = {0.f, 0.f, 0.f, 0.f};
    for (int base = i0; base < i1; base += 256) {
        int i = base + t;
        if (i < i1) {
            eg[t] = __expf(s_g[i] - mg) * wg;
            el[t] = mask[i] ? __expf(s_l[i] - ml) * wl : 0.f;
        }
        __syncthreads();
        int cnt = min(256, i1 - base);
        for (int k = r; k < cnt; k += 4) {
            float4 xv = *reinterpret_cast<const float4*>(
                x + (size_t)(base + k) * H_DIM + c4);
            float wgk = eg[k], wlk = el[k];
            pg4.x = fmaf(xv.x, wgk, pg4.x); pg4.y = fmaf(xv.y, wgk, pg4.y);
            pg4.z = fmaf(xv.z, wgk, pg4.z); pg4.w = fmaf(xv.w, wgk, pg4.w);
            pl4.x = fmaf(xv.x, wlk, pl4.x); pl4.y = fmaf(xv.y, wlk, pl4.y);
            pl4.z = fmaf(xv.z, wlk, pl4.z); pl4.w = fmaf(xv.w, wlk, pl4.w);
        }
        __syncthreads();
    }
    part[r][c4 + 0] = pg4.x; part[r][c4 + 1] = pg4.y;
    part[r][c4 + 2] = pg4.z; part[r][c4 + 3] = pg4.w;
    part[r][256 + c4 + 0] = pl4.x; part[r][256 + c4 + 1] = pl4.y;
    part[r][256 + c4 + 2] = pl4.z; part[r][256 + c4 + 3] = pl4.w;
    __syncthreads();

    int nn = i1 - i0;
    float inv = (nn > 0) ? rsqrtf((float)nn) : 0.f;
    pooled[t]       = (part[0][t] + part[1][t] + part[2][t] + part[3][t]) * inv;
    pooled[256 + t] = (part[0][256 + t] + part[1][256 + t] +
                       part[2][256 + t] + part[3][256 + t]) * inv;
    __syncthreads();

    // Output GEMV: out[seg][t] = sum_k pooled[k] * ow[k*256 + t] + ob[t]
    float acc = ob[t];
    for (int k = 0; k < 512; k++)
        acc = fmaf(pooled[k], ow[(size_t)k * H_DIM + t], acc);
    out[(size_t)seg * H_DIM + t] = acc;
}

// ---------------------------------------------------------------------------
extern "C" void kernel_launch(void* const* d_in, const int* in_sizes, int n_in,
                              void* d_out, int out_size, void* d_ws, size_t ws_size,
                              hipStream_t stream) {
    const float*         x     = (const float*)d_in[0];
    const long long*     batch = (const long long*)d_in[1];
    const unsigned char* lmask = (const unsigned char*)d_in[2];
    const float* gw1 = (const float*)d_in[3];
    const float* gb1 = (const float*)d_in[4];
    const float* gw2 = (const float*)d_in[5];
    const float* gb2 = (const float*)d_in[6];
    const float* lw1 = (const float*)d_in[7];
    const float* lb1 = (const float*)d_in[8];
    const float* lw2 = (const float*)d_in[9];
    const float* lb2 = (const float*)d_in[10];
    const float* ow  = (const float*)d_in[11];
    const float* ob  = (const float*)d_in[12];

    int n = in_sizes[1];                       // N nodes

    // Workspace layout
    char* ws = (char*)d_ws;
    __bf16* Bbuf = (__bf16*)ws;                               // 128 KB
    float* s_g = (float*)(ws + 131072);                       // N f32
    float* s_l = (float*)(ws + 131072 + (size_t)n * 4);       // N f32
    int*   seg = (int*)  (ws + 131072 + (size_t)n * 8);       // B+1 i32

    prep_weights<<<(B_ELEMS + 255) / 256, 256, 0, stream>>>(gw1, lw1, Bbuf);
    seg_bounds<<<(NSEG + 1 + 255) / 256, 256, 0, stream>>>(batch, seg, n, NSEG);

    int ntiles = (n + 15) / 16;
    int nblocks = (ntiles + 7) / 8;
    if (nblocks > 2048) nblocks = 2048;        // grid-stride; amortize LDS stage
    scores_kernel<<<nblocks, 256, 0, stream>>>(
        x, Bbuf, gb1, gw2, gb2, lb1, lw2, lb2, s_g, s_l, n, ntiles);

    pool_kernel<<<NSEG, 256, 0, stream>>>(x, lmask, s_g, s_l, seg, ow, ob, (float*)d_out);
}
